// DetectionHead_13400297963753
// MI455X (gfx1250) — compile-verified, matrix operand order checked by reading, not run on hardware
//
#include <hip/hip_runtime.h>
#include <hip/hip_fp16.h>
#include <stdint.h>
#include <float.h>

#define NLOC 21824
#define NB   16
#define NC   80
#define MDET 1000
#define CAP  4096
#define NBINS 4096
#define ROWS_PER_BLK 8
#define SCORE_THRESH 0.05f
#define IOU_THRESH   0.6f

typedef __attribute__((ext_vector_type(16))) _Float16 v16h;
typedef __attribute__((ext_vector_type(8)))  float    v8f;

struct LevelPtrs {
  const float* cls[5];
  const float* cnt[5];
  const float* reg[5];
};

// ---------------------------------------------------------------- utilities
__global__ void zero_kernel(float* p, int n) {
  int t = blockIdx.x * blockDim.x + threadIdx.x;
  if (t < n) p[t] = 0.0f;
}

// ---------------------------------------------------------------- pass 1: decode
// Bandwidth-bound: each thread owns one (b, location); 80-channel max over
// logits (sigmoid is monotone -> argmax/max on logits, one exp at the end).
__global__ void decode_kernel(LevelPtrs P, float* __restrict__ score,
                              float* __restrict__ clsv, float* __restrict__ box) {
  int t = blockIdx.x * blockDim.x + threadIdx.x;
  if (t >= NB * NLOC) return;
  int b = t / NLOC, i = t % NLOC;
  int lv, li;
  if      (i < 16384) { lv = 0; li = i; }
  else if (i < 20480) { lv = 1; li = i - 16384; }
  else if (i < 21504) { lv = 2; li = i - 20480; }
  else if (i < 21760) { lv = 3; li = i - 21504; }
  else                { lv = 4; li = i - 21760; }
  const int Wt[5] = {128, 64, 32, 16, 8};
  const int St[5] = {8, 16, 32, 64, 128};
  int W = Wt[lv], s = St[lv], HW = W * W;
  int h = li / W, w = li % W;
  float cx = (float)(w * s + (s >> 1));
  float cy = (float)(h * s + (s >> 1));

  const float* cp = P.cls[lv] + (size_t)b * NC * HW + li;
  float mx = -FLT_MAX; int am = 0;
  for (int c = 0; c < NC; ++c) {
    float v = cp[(size_t)c * HW];
    if (v > mx) { mx = v; am = c; }
  }
  float pc = 1.0f / (1.0f + __expf(-mx));
  float ct = P.cnt[lv][(size_t)b * HW + li];
  float pt = 1.0f / (1.0f + __expf(-ct));
  float sc = sqrtf(pc * pt);

  const float* rp = P.reg[lv] + (size_t)b * 4 * HW + li;
  float l  = rp[0];
  float tt = rp[(size_t)HW];
  float r  = rp[(size_t)2 * HW];
  float bb = rp[(size_t)3 * HW];

  size_t o = (size_t)b * NLOC + i;
  score[o] = sc;
  clsv[o]  = (float)(am + 1);
  box[o * 4 + 0] = cx - l;  box[o * 4 + 1] = cy - tt;
  box[o * 4 + 2] = cx + r;  box[o * 4 + 3] = cy + bb;
}

// ---------------------------------------------------------------- pass 2: histogram select
// Scores are in [0,1] => float bits are monotone. bin = bits>>18 (exp + 5
// mantissa bits, max 4064 < 4096). Find the bin holding the 1000th score.
__global__ void hist_kernel(const float* __restrict__ score, uint32_t* __restrict__ thrBin) {
  __shared__ uint32_t h[NBINS];
  int b = blockIdx.x;
  for (int k = threadIdx.x; k < NBINS; k += blockDim.x) h[k] = 0;
  __syncthreads();
  for (int i = threadIdx.x; i < NLOC; i += blockDim.x) {
    uint32_t bin = __float_as_uint(score[(size_t)b * NLOC + i]) >> 18;
    atomicAdd(&h[bin], 1u);
  }
  __syncthreads();
  if (threadIdx.x == 0) {
    uint32_t cum = 0; int T = 0;
    for (int k = NBINS - 1; k >= 0; --k) {
      if (cum + h[k] >= MDET) { T = k; break; }
      cum += h[k];
    }
    thrBin[b] = (uint32_t)T;
  }
}

__global__ void compact_kernel(const float* __restrict__ score,
                               const uint32_t* __restrict__ thrBin,
                               uint32_t* __restrict__ candCount,
                               uint32_t* __restrict__ candIdx) {
  int t = blockIdx.x * blockDim.x + threadIdx.x;
  if (t >= NB * NLOC) return;
  int b = t / NLOC, i = t % NLOC;
  uint32_t bin = __float_as_uint(score[(size_t)b * NLOC + i]) >> 18;
  if (bin >= thrBin[b]) {
    uint32_t pos = atomicAdd(&candCount[b], 1u);
    if (pos < CAP) candIdx[(size_t)b * CAP + pos] = i;
  }
}

// ---------------------------------------------------------------- pass 2b: padded gather
// Pack (score bits, original index) per candidate into an interleaved uint2
// array padded to CAP with (-FLT_MAX, INT_MAX), so the WMMA rank loop is a
// single unconditional coalesced 8B load per lane (no clamps/selects/branches).
__global__ void gather_kernel(const float* __restrict__ score,
                              const uint32_t* __restrict__ candCount,
                              const uint32_t* __restrict__ candIdx,
                              uint2* __restrict__ candSI) {
  int t = blockIdx.x * blockDim.x + threadIdx.x;
  if (t >= NB * CAP) return;
  int b = t / CAP, slot = t % CAP;
  uint32_t cc = candCount[b]; if (cc > CAP) cc = CAP;
  uint2 v;
  if ((uint32_t)slot < cc) {
    uint32_t ci = candIdx[(size_t)b * CAP + slot];
    v.x = __float_as_uint(score[(size_t)b * NLOC + ci]);
    v.y = ci;
  } else {
    v.x = __float_as_uint(-FLT_MAX);
    v.y = 0x7FFFFFFFu;
  }
  candSI[(size_t)b * CAP + slot] = v;
}

// ---------------------------------------------------------------- pass 3: WMMA exact rank
// One wave ranks 16 candidates against all candidates in K=32 tiles.
// Indicators go into 16x32 f16 A tiles (ISA 7.12.2 layout), B = ones,
// v_wmma_f32_16x16x32_f16 accumulates counts in f32 (exact; 16-bit XDL WMMA
// is TRANS-tracked so the matrix pipe co-executes with the VALU compares).
// Unrolled 2x with independent accumulators to keep the XDL pipe busy.
__global__ void __launch_bounds__(128)
rank_kernel(const float* __restrict__ clsv, const float* __restrict__ box,
            const uint32_t* __restrict__ candCount, const uint2* __restrict__ candSI,
            float* __restrict__ selScore, float* __restrict__ selClass,
            float* __restrict__ selBox) {
  const int lane = threadIdx.x & 31;
  const int wave = threadIdx.x >> 5;
  const int b    = blockIdx.x >> 6;                 // 64 blocks per batch
  const int grp  = (blockIdx.x & 63) * 4 + wave;    // group of 16 candidates
  uint32_t cc0 = candCount[b];
  const int cc = (int)(cc0 > CAP ? CAP : cc0);
  const int gbase = grp * 16;
  if (gbase >= cc) return;                          // wave-uniform

  const uint2* cs = candSI + (size_t)b * CAP;
  const int m = lane & 15;                          // my A-matrix row
  const int slot = gbase + m;                       // < CAP always
  uint2 mySI = cs[slot];
  float si  = __uint_as_float(mySI.x);
  int  iIdx = (int)mySI.y;
  const int validM = slot < cc;

  v16h bones;
#pragma unroll
  for (int e = 0; e < 16; ++e) bones[e] = (_Float16)1.0f;
  v8f acc0 = {}, acc1 = {};

  const int nIter = (cc + 63) >> 6;                 // 2 tiles (64 cands) per iter
  for (int it = 0; it < nIter; ++it) {
    uint2 j0 = cs[it * 64 + lane];                  // padded => always in-bounds
    uint2 j1 = cs[it * 64 + 32 + lane];
    float sj0 = __uint_as_float(j0.x);  int jx0 = (int)j0.y;
    float sj1 = __uint_as_float(j1.x);  int jx1 = (int)j1.y;

    v16h A0, A1;
#pragma unroll
    for (int e = 0; e < 16; ++e) {
      // 16-bit A 16x32 layout: lanes0-15 e0-7->K0-7, e8-15->K16-23;
      //                        lanes16-31 e0-7->K8-15, e8-15->K24-31.
      int K = (e < 8) ? ((lane < 16) ? e : e + 8)
                      : ((lane < 16) ? e + 8 : e + 16);
      float s0 = __shfl(sj0, K, 32); int x0 = __shfl(jx0, K, 32);
      float s1 = __shfl(sj1, K, 32); int x1 = __shfl(jx1, K, 32);
      bool i0 = (s0 > si) || ((s0 == si) && (x0 < iIdx));
      bool i1 = (s1 > si) || ((s1 == si) && (x1 < iIdx));
      A0[e] = i0 ? (_Float16)1.0f : (_Float16)0.0f;
      A1[e] = i1 ? (_Float16)1.0f : (_Float16)0.0f;
    }
    acc0 = __builtin_amdgcn_wmma_f32_16x16x32_f16(false, A0, false, bones,
                                                  (short)0, acc0, false, false);
    acc1 = __builtin_amdgcn_wmma_f32_16x16x32_f16(false, A1, false, bones,
                                                  (short)0, acc1, false, false);
  }

  // C layout: VGPR e holds M=e (lanes 0-15) and M=e+8 (lanes 16-31).
  float myRank = 0.0f;
#pragma unroll
  for (int e = 0; e < 8; ++e) {
    float r = acc0[e] + acc1[e];
    float rlo = __shfl(r, 0, 32);
    float rhi = __shfl(r, 16, 32);
    if (lane == e)     myRank = rlo;
    if (lane == e + 8) myRank = rhi;
  }
  if (lane < 16 && validM) {
    int r = (int)(myRank + 0.5f);
    if (r < MDET) {
      size_t so = (size_t)b * MDET + r;
      selScore[so] = si;
      selClass[so] = clsv[(size_t)b * NLOC + iIdx];
#pragma unroll
      for (int k = 0; k < 4; ++k)
        selBox[so * 4 + k] = box[((size_t)b * NLOC + iIdx) * 4 + k];
    }
  }
}

// ---------------------------------------------------------------- pass 4: per-image box max
__global__ void boxmax_kernel(const float* __restrict__ selBox, float* __restrict__ boxMax) {
  __shared__ float red[256];
  int b = blockIdx.x;
  float mx = -FLT_MAX;
  for (int i = threadIdx.x; i < MDET * 4; i += blockDim.x)
    mx = fmaxf(mx, selBox[(size_t)b * MDET * 4 + i]);
  red[threadIdx.x] = mx;
  __syncthreads();
  for (int s = 128; s > 0; s >>= 1) {
    if (threadIdx.x < s) red[threadIdx.x] = fmaxf(red[threadIdx.x], red[threadIdx.x + s]);
    __syncthreads();
  }
  if (threadIdx.x == 0) boxMax[b] = red[0];
}

// ---------------------------------------------------------------- pass 5: suppression matrix
// Stage the 1000 selected boxes (16KB) into LDS with CDNA5 async-to-LDS
// copies (ASYNCcnt path), apply class offsets in place, then each thread
// emits one 32-bit suppression word.
__global__ void __launch_bounds__(256)
iou_kernel(const float* __restrict__ selBox, const float* __restrict__ selClass,
           const float* __restrict__ boxMax, uint32_t* __restrict__ supmat) {
  __shared__ float4 sbox[MDET];
  int b     = blockIdx.x / (MDET / ROWS_PER_BLK);
  int chunk = blockIdx.x % (MDET / ROWS_PER_BLK);
  const float* src = selBox + (size_t)b * MDET * 4;

  for (int t = threadIdx.x; t < MDET; t += blockDim.x) {
    uint32_t ldsoff = (uint32_t)(uintptr_t)(&sbox[t]);   // flat LDS addr low bits = LDS offset
    uint32_t voff   = (uint32_t)(t * 16);
    asm volatile("global_load_async_to_lds_b128 %0, %1, %2"
                 :: "v"(ldsoff), "v"(voff), "s"(src) : "memory");
  }
  asm volatile("s_wait_asynccnt 0" ::: "memory");
  __syncthreads();

  float mx1 = boxMax[b] + 1.0f;
  for (int t = threadIdx.x; t < MDET; t += blockDim.x) {
    float off = selClass[(size_t)b * MDET + t] * mx1;
    float4 v = sbox[t];
    v.x += off; v.y += off; v.z += off; v.w += off;
    sbox[t] = v;
  }
  __syncthreads();

  int i = chunk * ROWS_PER_BLK + (threadIdx.x >> 5);
  int w = threadIdx.x & 31;
  float4 bi = sbox[i];
  float ai = (bi.z - bi.x + 1.0f) * (bi.w - bi.y + 1.0f);
  uint32_t bits = 0;
#pragma unroll 4
  for (int jj = 0; jj < 32; ++jj) {
    int j = w * 32 + jj;
    if (j < MDET && j > i) {
      float4 bj = sbox[j];
      float aj  = (bj.z - bj.x + 1.0f) * (bj.w - bj.y + 1.0f);
      float xx1 = fmaxf(bi.x, bj.x), yy1 = fmaxf(bi.y, bj.y);
      float xx2 = fminf(bi.z, bj.z), yy2 = fminf(bi.w, bj.w);
      float inter = fmaxf(xx2 - xx1, 0.0f) * fmaxf(yy2 - yy1, 0.0f);
      float iou = inter / (ai + aj - inter);
      if (iou > IOU_THRESH) bits |= (1u << jj);
    }
  }
  supmat[((size_t)b * MDET + i) * 32 + w] = bits;
}

// ---------------------------------------------------------------- pass 6: wave32 NMS scan + output
__global__ void nms_kernel(const float* __restrict__ selScore, const float* __restrict__ selClass,
                           const float* __restrict__ selBox, const uint32_t* __restrict__ supmat,
                           float* __restrict__ outS, float* __restrict__ outC,
                           float* __restrict__ outB) {
  int b = blockIdx.x;
  int lane = threadIdx.x;        // 32 threads = 1 wave; lane holds keep word `lane`
  uint32_t keep = 0;
  for (int jj = 0; jj < 32; ++jj) {
    int j = lane * 32 + jj;
    if (j < MDET && selScore[(size_t)b * MDET + j] >= SCORE_THRESH) keep |= (1u << jj);
  }
  for (int i = 0; i < MDET; ++i) {
    uint32_t kw = (uint32_t)__shfl((int)keep, i >> 5, 32);   // wave-uniform
    if ((kw >> (i & 31)) & 1u) {
      uint32_t sup = supmat[((size_t)b * MDET + i) * 32 + lane];
      keep &= ~sup;
    }
  }
  for (int jj = 0; jj < 32; ++jj) {
    int j = lane * 32 + jj;
    if (j < MDET) {
      float kf = ((keep >> jj) & 1u) ? 1.0f : 0.0f;
      size_t o = (size_t)b * MDET + j;
      outS[o] = selScore[o] * kf;
      outC[o] = selClass[o] * kf;
#pragma unroll
      for (int k = 0; k < 4; ++k) outB[o * 4 + k] = selBox[o * 4 + k] * kf;
    }
  }
}

// ---------------------------------------------------------------- host launch
extern "C" void kernel_launch(void* const* d_in, const int* in_sizes, int n_in,
                              void* d_out, int out_size, void* d_ws, size_t ws_size,
                              hipStream_t stream) {
  (void)in_sizes; (void)n_in; (void)ws_size;
  LevelPtrs P;
  for (int l = 0; l < 5; ++l) {
    P.cls[l] = (const float*)d_in[3 * l + 0];
    P.cnt[l] = (const float*)d_in[3 * l + 1];
    P.reg[l] = (const float*)d_in[3 * l + 2];
  }
  float* ws      = (float*)d_ws;
  float* score   = ws;
  float* clsv    = score + (size_t)NB * NLOC;
  float* box     = clsv + (size_t)NB * NLOC;
  float* after   = box + (size_t)NB * NLOC * 4;
  uint32_t* thrBin    = (uint32_t*)after;
  uint32_t* candCount = thrBin + NB;
  uint32_t* candIdx   = candCount + NB;
  uint2*    candSI    = (uint2*)(candIdx + (size_t)NB * CAP);
  float* selScore = (float*)(candSI + (size_t)NB * CAP);
  float* selClass = selScore + NB * MDET;
  float* selBox   = selClass + NB * MDET;
  float* boxMax   = selBox + (size_t)NB * MDET * 4;
  uint32_t* supmat = (uint32_t*)(boxMax + NB);

  float* outS = (float*)d_out;
  float* outC = outS + NB * MDET;
  float* outB = outC + NB * MDET;

  // per-call zero init (outputs + scatter targets + atomic counters)
  zero_kernel<<<1, 256, 0, stream>>>((float*)candCount, NB);
  int selN = 6 * NB * MDET;     // selScore+selClass+selBox contiguous
  zero_kernel<<<(selN + 255) / 256, 256, 0, stream>>>(selScore, selN);
  zero_kernel<<<(out_size + 255) / 256, 256, 0, stream>>>((float*)d_out, out_size);

  int tot = NB * NLOC;
  decode_kernel<<<(tot + 255) / 256, 256, 0, stream>>>(P, score, clsv, box);
  hist_kernel<<<NB, 256, 0, stream>>>(score, thrBin);
  compact_kernel<<<(tot + 255) / 256, 256, 0, stream>>>(score, thrBin, candCount, candIdx);
  gather_kernel<<<(NB * CAP + 255) / 256, 256, 0, stream>>>(score, candCount, candIdx, candSI);
  rank_kernel<<<NB * 64, 128, 0, stream>>>(clsv, box, candCount, candSI,
                                           selScore, selClass, selBox);
  boxmax_kernel<<<NB, 256, 0, stream>>>(selBox, boxMax);
  iou_kernel<<<NB * (MDET / ROWS_PER_BLK), 256, 0, stream>>>(selBox, selClass, boxMax, supmat);
  nms_kernel<<<NB, 32, 0, stream>>>(selScore, selClass, selBox, supmat, outS, outC, outB);
}
